// Separateq0qgqkckMultiplePassGNNReachabilityNet_67138928771753
// MI455X (gfx1250) — compile-verified, matrix operand order checked by reading, not run on hardware
//
#include <hip/hip_runtime.h>
#include <hip/hip_bf16.h>

typedef __attribute__((ext_vector_type(16))) _Float16 v16h;
typedef __attribute__((ext_vector_type(8)))  float    v8f;

// slope 0.01 < 1  =>  leaky_relu(x) == max(x, 0.01x)  (v_mul + v_max_num)
__device__ __forceinline__ float lrelu(float x) { return fmaxf(x, 0.01f * x); }

__device__ __forceinline__ void atomic_add_f32(float* p, float v) {
#if defined(__HIP_DEVICE_COMPILE__)
    unsafeAtomicAdd(p, v);   // native global_atomic_add_f32
#endif
}

// K-slot held by (lane, half h) in the 16-bit A/B WMMA operand layout:
// lanes 0-15:  halves 0..7 -> K 0..7,  halves 8..15 -> K 16..23
// lanes 16-31: halves 0..7 -> K 8..15, halves 8..15 -> K 24..31
__device__ __forceinline__ int kmap(int lane, int h) {
    return h + ((lane & 16) ? 8 : 0) + ((h & 8) ? 8 : 0);
}

__device__ __forceinline__ v8f wmma16(v16h a, v16h b, v8f c) {
    return __builtin_amdgcn_wmma_f32_16x16x32_f16(false, a, false, b, (short)0, c, false, false);
}

// ---------------- packed weight tiles ----------------
// Each tile = exact WMMA A-operand register image: 32 lanes x 16 halfs = 512 halfs.
enum {
    T_CW1 = 0,   // 2 tiles  (32x24)
    T_CW2 = 2,   // 2 tiles  (32x32)
    T_VW1 = 4,   // 4 tiles  (32x64: 2 row-tiles x 2 K-chunks)
    T_VW2 = 8,   // 2 tiles
    T_EW1 = 10,  // 6 tiles  (32x68: 2 row-tiles x 3 K-chunks)
    T_EW2 = 16,  // 2 tiles
    T_AW1 = 18,  // 2 tiles
    T_AW2 = 20,  // 2 tiles
    T_OW1 = 22,  // 2 tiles
    T_COUNT = 24
};

__device__ __forceinline__ void pack_store(_Float16* __restrict__ dst,
                                           const float* __restrict__ W, int in_ch,
                                           int tile, int k_off, int lane) {
    _Float16* p = dst + lane * 16;
    int m = tile * 16 + (lane & 15);
#pragma unroll
    for (int h = 0; h < 16; ++h) {
        int k = k_off + kmap(lane, h);
        p[h] = (k < in_ch) ? (_Float16)W[m * in_ch + k] : (_Float16)0.f;
    }
}

__global__ void k_pack(const float* cW1, const float* cW2, const float* vW1,
                       const float* vW2, const float* eW1, const float* eW2,
                       const float* aW1, const float* aW2, const float* oW1,
                       _Float16* __restrict__ wp) {
    int lane = threadIdx.x & 31;
    pack_store(wp + (T_CW1 + 0) * 512, cW1, 24, 0, 0, lane);
    pack_store(wp + (T_CW1 + 1) * 512, cW1, 24, 1, 0, lane);
    pack_store(wp + (T_CW2 + 0) * 512, cW2, 32, 0, 0, lane);
    pack_store(wp + (T_CW2 + 1) * 512, cW2, 32, 1, 0, lane);
#pragma unroll
    for (int t = 0; t < 2; ++t)
#pragma unroll
        for (int kc = 0; kc < 2; ++kc)
            pack_store(wp + (T_VW1 + t * 2 + kc) * 512, vW1, 64, t, kc * 32, lane);
    pack_store(wp + (T_VW2 + 0) * 512, vW2, 32, 0, 0, lane);
    pack_store(wp + (T_VW2 + 1) * 512, vW2, 32, 1, 0, lane);
#pragma unroll
    for (int t = 0; t < 2; ++t)
#pragma unroll
        for (int kc = 0; kc < 3; ++kc)
            pack_store(wp + (T_EW1 + t * 3 + kc) * 512, eW1, 68, t, kc * 32, lane);
    pack_store(wp + (T_EW2 + 0) * 512, eW2, 32, 0, 0, lane);
    pack_store(wp + (T_EW2 + 1) * 512, eW2, 32, 1, 0, lane);
    pack_store(wp + (T_AW1 + 0) * 512, aW1, 32, 0, 0, lane);
    pack_store(wp + (T_AW1 + 1) * 512, aW1, 32, 1, 0, lane);
    pack_store(wp + (T_AW2 + 0) * 512, aW2, 32, 0, 0, lane);
    pack_store(wp + (T_AW2 + 1) * 512, aW2, 32, 1, 0, lane);
    pack_store(wp + (T_OW1 + 0) * 512, oW1, 32, 0, 0, lane);
    pack_store(wp + (T_OW1 + 1) * 512, oW1, 32, 1, 0, lane);
}

// one 32-byte vector load per A tile (works for global or LDS base)
__device__ __forceinline__ v16h load_pack(const _Float16* wp, int tile, int lane) {
    return *(const v16h*)(wp + ((tile * 32 + lane) << 4));
}

// cooperative copy of n_tiles packed tiles (1 KB each) into LDS
__device__ __forceinline__ void stage_weights(_Float16* dst_lds,
                                              const _Float16* __restrict__ src,
                                              int n_tiles) {
    const uint4* s = (const uint4*)src;
    uint4* d = (uint4*)dst_lds;
    int n = n_tiles * 64;                    // 16B chunks
    for (int i = threadIdx.x; i < n; i += blockDim.x) d[i] = s[i];
}

// C/D tile element j on this lane is out-channel tile*16 + j + (lane>=16 ? 8 : 0)
__device__ __forceinline__ void bias_act(v8f& c, const float* __restrict__ bias,
                                         int tile, int lane) {
    int off = tile * 16 + ((lane & 16) ? 8 : 0);
#pragma unroll
    for (int j = 0; j < 8; ++j) c[j] = lrelu(c[j] + bias[off + j]);
}

// Repack two C tiles (ch 0-15, 16-31) into next layer's B operand (lane-local).
__device__ __forceinline__ v16h repack(v8f c0, v8f c1) {
    v16h b;
#pragma unroll
    for (int h = 0; h < 8; ++h) {
        b[h]     = (_Float16)c0[h];
        b[h + 8] = (_Float16)c1[h];
    }
    return b;
}

__device__ __forceinline__ v8f zero8() {
    v8f z = {0.f, 0.f, 0.f, 0.f, 0.f, 0.f, 0.f, 0.f};
    return z;
}

// node-major activation row: 34 used + 6 pad halfs = 80 bytes (16B aligned)
#define ROW_ACT 40
// node-major new_vertex f16 mirror: 32 halfs = 64 bytes
#define ROW_NV  32

union Row40 { float4 f4[5]; _Float16 h[ROW_ACT]; };
union Row32 { float4 f4[4]; _Float16 h[ROW_NV]; };

// Build B operand for a 32-channel node-major f16 row (compile-time indices).
__device__ __forceinline__ v16h packB_row32(const Row32& r, int lane) {
    v16h x;
    if (lane < 16) {
#pragma unroll
        for (int h = 0; h < 8; ++h) { x[h] = r.h[h];      x[h + 8] = r.h[16 + h]; }
    } else {
#pragma unroll
        for (int h = 0; h < 8; ++h) { x[h] = r.h[8 + h];  x[h + 8] = r.h[24 + h]; }
    }
    return x;
}

// ---- cfg branch: 24-feature axis front + cfg MLP (24->32->32) ----
// 'wt' points at LDS copy of tiles T_CW1..T_CW2+1 (local indices 0..3)
__device__ __forceinline__ void cfg_branch(
    float k0, float k1, float k2, float o0, float o1, float o2,
    const float* xW, const float* xb, const float* yW, const float* yb,
    const float* tW, const float* tb,
    const _Float16* wt, const float* cb1, const float* cb2,
    int lane, v8f& out0, v8f& out1) {
    v16h fB;
#pragma unroll
    for (int h = 0; h < 16; ++h) {
        int k = kmap(lane, h);
        float val = 0.f;
        if (k < 24) {
            int axis = k >> 3, j = k & 7;
            const float* W  = (axis == 0) ? xW : (axis == 1) ? yW : tW;
            const float* bb = (axis == 0) ? xb : (axis == 1) ? yb : tb;
            float a = (axis == 0) ? k0 : (axis == 1) ? k1 : k2;
            float o = (axis == 0) ? o0 : (axis == 1) ? o1 : o2;
            val = lrelu(W[j * 2 + 0] * a + W[j * 2 + 1] * o + bb[j]);
        }
        fB[h] = (_Float16)val;
    }
    v8f z = zero8();
    v8f h0 = wmma16(load_pack(wt, 0, lane), fB, z);
    v8f h1 = wmma16(load_pack(wt, 1, lane), fB, z);
    bias_act(h0, cb1, 0, lane);
    bias_act(h1, cb1, 1, lane);
    v16h hB = repack(h0, h1);
    out0 = wmma16(load_pack(wt, 2, lane), hB, z);
    out1 = wmma16(load_pack(wt, 3, lane), hB, z);
    bias_act(out0, cb2, 0, lane);
    bias_act(out1, cb2, 1, lane);
}

// ================= Kernel 1: per-node front (cfg x2 + vtx MLP) =================
__global__ void k_front(const float* __restrict__ verts,
                        const float* xW, const float* xb,
                        const float* yW, const float* yb,
                        const float* tW, const float* tb,
                        const float* cb1, const float* cb2,
                        const float* vb1, const float* vb2,
                        const _Float16* __restrict__ wp,
                        _Float16* __restrict__ vout,
                        int B, int N) {
    __shared__ _Float16 sw[10 * 512];            // tiles T_CW1..T_VW2+1
    stage_weights(sw, wp + T_CW1 * 512, 10);
    __syncthreads();

    int lane = threadIdx.x & 31;
    int wave = threadIdx.x >> 5;
    int tile = blockIdx.x * (blockDim.x >> 5) + wave;
    int total = B * N;
    int col0 = tile * 16;
    if (col0 >= total) return;                   // wave-uniform exit
    int col = col0 + (lane & 15);
    bool active = col < total;
    if (!active) col = total - 1;
    const float* vp = verts + (long)col * 11;
    float qk0 = vp[0], qk1 = vp[1], qk2 = vp[2];
    float q00 = vp[3], q01 = vp[4], q02 = vp[5];
    float qg0 = vp[6], qg1 = vp[7], qg2 = vp[8];
    float c9 = vp[9], c10 = vp[10];

    v8f a0, a1, b0, b1;
    cfg_branch(qk0, qk1, qk2, q00, q01, q02, xW, xb, yW, yb, tW, tb,
               sw, cb1, cb2, lane, a0, a1);
    cfg_branch(qk0, qk1, qk2, qg0, qg1, qg2, xW, xb, yW, yb, tW, tb,
               sw, cb1, cb2, lane, b0, b1);
    v16h cfgB0 = repack(a0, a1);                 // config channels 0..31
    v16h cfgB1 = repack(b0, b1);                 // config channels 32..63

    v8f z = zero8();
    v8f v0 = wmma16(load_pack(sw, T_VW1 + 0, lane), cfgB0, z);
    v0     = wmma16(load_pack(sw, T_VW1 + 1, lane), cfgB1, v0);
    v8f v1 = wmma16(load_pack(sw, T_VW1 + 2, lane), cfgB0, z);
    v1     = wmma16(load_pack(sw, T_VW1 + 3, lane), cfgB1, v1);
    bias_act(v0, vb1, 0, lane);
    bias_act(v1, vb1, 1, lane);
    v16h vB = repack(v0, v1);
    v8f w0 = wmma16(load_pack(sw, T_VW2 + 0, lane), vB, z);
    v8f w1 = wmma16(load_pack(sw, T_VW2 + 1, lane), vB, z);
    bias_act(w0, vb2, 0, lane);
    bias_act(w1, vb2, 1, lane);

    if (active) {
        int choff = (lane & 16) ? 8 : 0;
        _Float16* ob = vout + (long)col * ROW_ACT;
#pragma unroll
        for (int j = 0; j < 8; ++j) {
            ob[choff + j]      = (_Float16)w0[j];
            ob[16 + choff + j] = (_Float16)w1[j];
        }
        ob[32] = (_Float16)c9;                   // col channels (dup store from both
        ob[33] = (_Float16)c10;                  // lane halves; same value, harmless)
    }
}

// ================= Kernel 2: per-edge message MLP + scatter add ================
__global__ void k_edge(const _Float16* __restrict__ vf,
                       const int* __restrict__ edges,
                       const float* eb1, const float* eb2,
                       const _Float16* __restrict__ wp,
                       float* __restrict__ sums,
                       int B, int N, int E) {
    __shared__ _Float16 sw[8 * 512];             // tiles T_EW1..T_EW2+1
    stage_weights(sw, wp + T_EW1 * 512, 8);
    __syncthreads();

    int lane = threadIdx.x & 31;
    int wave = threadIdx.x >> 5;
    int tile = blockIdx.x * (blockDim.x >> 5) + wave;
    int total = B * E;
    int col0 = tile * 16;
    if (col0 >= total) return;
    int col = col0 + (lane & 15);
    bool active = col < total;
    if (!active) col = total - 1;
    int b = col / E, e = col % E;
    int s = edges[e], d = edges[E + e];

    // vector-load the two 80B node rows
    Row40 S, D;
    const float4* sp = (const float4*)(vf + ((long)b * N + s) * ROW_ACT);
    const float4* dp = (const float4*)(vf + ((long)b * N + d) * ROW_ACT);
#pragma unroll
    for (int i = 0; i < 5; ++i) { S.f4[i] = sp[i]; D.f4[i] = dp[i]; }

    // Assemble the three K=32 chunks of the 68-feature vector
    // (feature k: k<34 -> src ch k, else dst ch k-34) with constant indices.
    v16h f0, f1, f2;
    if (lane < 16) {
#pragma unroll
        for (int h = 0; h < 8; ++h) { f0[h] = S.h[h];     f0[h + 8] = S.h[16 + h]; }
        f1[0] = S.h[32]; f1[1] = S.h[33];
#pragma unroll
        for (int h = 2; h < 8; ++h)  f1[h] = D.h[h - 2];
#pragma unroll
        for (int h = 0; h < 8; ++h)  f1[h + 8] = D.h[14 + h];
#pragma unroll
        for (int h = 0; h < 4; ++h)  f2[h] = D.h[30 + h];
#pragma unroll
        for (int h = 4; h < 16; ++h) f2[h] = (_Float16)0.f;
    } else {
#pragma unroll
        for (int h = 0; h < 8; ++h) { f0[h] = S.h[8 + h]; f0[h + 8] = S.h[24 + h]; }
#pragma unroll
        for (int h = 0; h < 8; ++h) { f1[h] = D.h[6 + h]; f1[h + 8] = D.h[22 + h]; }
#pragma unroll
        for (int h = 0; h < 16; ++h) f2[h] = (_Float16)0.f;
    }

    v8f z = zero8();
    v8f m0 = wmma16(load_pack(sw, 0, lane), f0, z);   // EW1 t0 k0
    m0     = wmma16(load_pack(sw, 1, lane), f1, m0);  // EW1 t0 k32
    m0     = wmma16(load_pack(sw, 2, lane), f2, m0);  // EW1 t0 k64
    v8f m1 = wmma16(load_pack(sw, 3, lane), f0, z);   // EW1 t1 k0
    m1     = wmma16(load_pack(sw, 4, lane), f1, m1);  // EW1 t1 k32
    m1     = wmma16(load_pack(sw, 5, lane), f2, m1);  // EW1 t1 k64
    bias_act(m0, eb1, 0, lane);
    bias_act(m1, eb1, 1, lane);
    v16h mB = repack(m0, m1);
    v8f o0 = wmma16(load_pack(sw, 6, lane), mB, z);   // EW2 t0
    v8f o1 = wmma16(load_pack(sw, 7, lane), mB, z);   // EW2 t1
    bias_act(o0, eb2, 0, lane);
    bias_act(o1, eb2, 1, lane);

    if (active) {
        int choff = (lane & 16) ? 8 : 0;
        float* sb = sums + (long)b * 32 * N;
#pragma unroll
        for (int j = 0; j < 8; ++j) {
            float u0 = o0[j], u1 = o1[j];
            int ca = choff + j, cb = 16 + choff + j;
            atomic_add_f32(sb + (long)ca * N + d, u0);
            atomic_add_f32(sb + (long)ca * N + s, u0);
            atomic_add_f32(sb + (long)cb * N + d, u1);
            atomic_add_f32(sb + (long)cb * N + s, u1);
        }
    }
}

// ====== Kernel 3: scatter-mean finalize (set/accumulate) + f16 node-major mirror
__global__ void k_finalize(const float* __restrict__ sums,
                           const float* __restrict__ icnt,   // 1/cnt (0 if cnt==0)
                           float* __restrict__ nv,
                           _Float16* __restrict__ nvh,
                           int N, int total, int setmode) {
    int i = blockIdx.x * blockDim.x + threadIdx.x;
    if (i >= total) return;
    int n = i % N;
    int bc = i / N;               // b*32 + c
    int c = bc & 31, b = bc >> 5;
    float v = sums[i] * icnt[n];
    float nvv = setmode ? v : (nv[i] + v);
    nv[i] = nvv;
    nvh[((long)b * N + n) * ROW_NV + c] = (_Float16)nvv;
}

// ================= Kernel 4: per-node aft MLP (32->32->32) + col ==============
__global__ void k_aft(const _Float16* __restrict__ nvh,
                      const float* __restrict__ verts,
                      const float* ab1, const float* ab2,
                      const _Float16* __restrict__ wp,
                      _Float16* __restrict__ hout,
                      int B, int N) {
    __shared__ _Float16 sw[4 * 512];             // tiles T_AW1..T_AW2+1
    stage_weights(sw, wp + T_AW1 * 512, 4);
    __syncthreads();

    int lane = threadIdx.x & 31;
    int wave = threadIdx.x >> 5;
    int tile = blockIdx.x * (blockDim.x >> 5) + wave;
    int total = B * N;
    int col0 = tile * 16;
    if (col0 >= total) return;
    int col = col0 + (lane & 15);
    bool active = col < total;
    if (!active) col = total - 1;

    Row32 r;
    const float4* rp = (const float4*)(nvh + (long)col * ROW_NV);
#pragma unroll
    for (int i = 0; i < 4; ++i) r.f4[i] = rp[i];
    v16h xB = packB_row32(r, lane);

    v8f z = zero8();
    v8f h0 = wmma16(load_pack(sw, 0, lane), xB, z);
    v8f h1 = wmma16(load_pack(sw, 1, lane), xB, z);
    bias_act(h0, ab1, 0, lane);
    bias_act(h1, ab1, 1, lane);
    v16h hB = repack(h0, h1);
    v8f g0 = wmma16(load_pack(sw, 2, lane), hB, z);
    v8f g1 = wmma16(load_pack(sw, 3, lane), hB, z);
    bias_act(g0, ab2, 0, lane);
    bias_act(g1, ab2, 1, lane);

    if (active) {
        int choff = (lane & 16) ? 8 : 0;
        _Float16* ob = hout + (long)col * ROW_ACT;
#pragma unroll
        for (int j = 0; j < 8; ++j) {
            ob[choff + j]      = (_Float16)g0[j];
            ob[16 + choff + j] = (_Float16)g1[j];
        }
        const float* vp = verts + (long)col * 11;
        ob[32] = (_Float16)vp[9];
        ob[33] = (_Float16)vp[10];
    }
}

// ================= Kernel 5: vo head (32->32->1) =================
__global__ void k_vo(const _Float16* __restrict__ nvh,
                     const float* ob1, const float* oW2, const float* ob2,
                     const _Float16* __restrict__ wp,
                     float* __restrict__ fv, int B, int N) {
    __shared__ _Float16 sw[2 * 512];             // tiles T_OW1..T_OW1+1
    stage_weights(sw, wp + T_OW1 * 512, 2);
    __syncthreads();

    int lane = threadIdx.x & 31;
    int wave = threadIdx.x >> 5;
    int tile = blockIdx.x * (blockDim.x >> 5) + wave;
    int total = B * N;
    int col0 = tile * 16;
    if (col0 >= total) return;
    int col = col0 + (lane & 15);
    bool active = col < total;
    if (!active) col = total - 1;

    Row32 r;
    const float4* rp = (const float4*)(nvh + (long)col * ROW_NV);
#pragma unroll
    for (int i = 0; i < 4; ++i) r.f4[i] = rp[i];
    v16h xB = packB_row32(r, lane);

    v8f z = zero8();
    v8f h0 = wmma16(load_pack(sw, 0, lane), xB, z);
    v8f h1 = wmma16(load_pack(sw, 1, lane), xB, z);
    bias_act(h0, ob1, 0, lane);
    bias_act(h1, ob1, 1, lane);

    // vo_W2 is 1x32: per-lane partial dot over this lane's channel set,
    // then combine the two lane-halves of the column via xor-16 shuffle.
    int choff = (lane & 16) ? 8 : 0;
    float p = 0.f;
#pragma unroll
    for (int j = 0; j < 8; ++j)
        p += oW2[choff + j] * h0[j] + oW2[16 + choff + j] * h1[j];
    p += __shfl_xor(p, 16, 32);
    if (active && lane < 16)
        fv[col] = lrelu(p + ob2[0]);
}

// ================= Kernel 6: per-batch reduction + sigmoid =================
__global__ void k_out(const float* __restrict__ fv,
                      const float* __restrict__ goW,
                      const float* __restrict__ gob,
                      float* __restrict__ out, int N) {
    __shared__ float red[256];
    int b = blockIdx.x;
    float s = 0.f;
    for (int n = threadIdx.x; n < N; n += blockDim.x)
        s += fv[(long)b * N + n] * goW[n];
    red[threadIdx.x] = s;
    __syncthreads();
    for (int off = blockDim.x >> 1; off > 0; off >>= 1) {
        if ((int)threadIdx.x < off) red[threadIdx.x] += red[threadIdx.x + off];
        __syncthreads();
    }
    if (threadIdx.x == 0) {
        float x = red[0] + gob[0];
        out[b] = 1.f / (1.f + expf(-x));
    }
}

// ================= utility kernels =================
__global__ void k_zero(float* __restrict__ p, int n) {
    int i = blockIdx.x * blockDim.x + threadIdx.x;
    if (i < n) p[i] = 0.f;
}

__global__ void k_count(const int* __restrict__ edges, float* __restrict__ cnt, int E) {
    int e = blockIdx.x * blockDim.x + threadIdx.x;
    if (e < E) {
        atomic_add_f32(&cnt[edges[e]], 1.f);       // src
        atomic_add_f32(&cnt[edges[E + e]], 1.f);   // dst
    }
}

// icnt[n] = cnt[n] > 0 ? 1/cnt[n] : 0   (hoists the divide out of k_finalize)
__global__ void k_invcnt(const float* __restrict__ cnt, float* __restrict__ icnt, int N) {
    int i = blockIdx.x * blockDim.x + threadIdx.x;
    if (i < N) {
        float c = cnt[i];
        icnt[i] = (c > 0.f) ? __frcp_rn(c) : 0.f;
    }
}

extern "C" void kernel_launch(void* const* d_in, const int* in_sizes, int n_in,
                              void* d_out, int out_size, void* d_ws, size_t ws_size,
                              hipStream_t stream) {
    const float* verts = (const float*)d_in[0];
    const int*   edges = (const int*)d_in[1];
    const float* xW = (const float*)d_in[2],  *xb = (const float*)d_in[3];
    const float* yW = (const float*)d_in[4],  *yb = (const float*)d_in[5];
    const float* tW = (const float*)d_in[6],  *tb = (const float*)d_in[7];
    const float* cW1 = (const float*)d_in[8], *cb1 = (const float*)d_in[9];
    const float* cW2 = (const float*)d_in[10],*cb2 = (const float*)d_in[11];
    const float* vW1 = (const float*)d_in[12],*vb1 = (const float*)d_in[13];
    const float* vW2 = (const float*)d_in[14],*vb2 = (const float*)d_in[15];
    const float* eW1 = (const float*)d_in[16],*eb1 = (const float*)d_in[17];
    const float* eW2 = (const float*)d_in[18],*eb2 = (const float*)d_in[19];
    const float* aW1 = (const float*)d_in[20],*ab1 = (const float*)d_in[21];
    const float* aW2 = (const float*)d_in[22],*ab2 = (const float*)d_in[23];
    const float* oW1 = (const float*)d_in[24],*ob1 = (const float*)d_in[25];
    const float* oW2 = (const float*)d_in[26],*ob2 = (const float*)d_in[27];
    const float* goW = (const float*)d_in[28],*gob = (const float*)d_in[29];

    int N = in_sizes[28];               // go_W has shape (1, N_NODES)
    int E = in_sizes[1] / 2;
    int B = in_sizes[0] / (N * 11);

    // workspace carve-out
    char* w = (char*)d_ws;
    auto carve = [&](size_t bytes) {
        void* p = (void*)w;
        w += (bytes + 255) & ~(size_t)255;
        return p;
    };
    _Float16* wpk  = (_Float16*)carve((size_t)T_COUNT * 512 * sizeof(_Float16));
    _Float16* v_h  = (_Float16*)carve((size_t)B * N * ROW_ACT * sizeof(_Float16));
    _Float16* h_h  = (_Float16*)carve((size_t)B * N * ROW_ACT * sizeof(_Float16));
    _Float16* nvh  = (_Float16*)carve((size_t)B * N * ROW_NV * sizeof(_Float16));
    float*    nv   = (float*)carve((size_t)B * 32 * N * sizeof(float));
    float*    sums = (float*)carve((size_t)B * 32 * N * sizeof(float));
    float*    cnt  = (float*)carve((size_t)N * sizeof(float));
    float*    icnt = (float*)carve((size_t)N * sizeof(float));
    float*    fv   = (float*)carve((size_t)B * N * sizeof(float));

    const int T = 256;                  // 8 waves per block
    int tiles_v  = (B * N + 15) / 16;
    int blocks_v = (tiles_v + 7) / 8;
    int tiles_e  = (B * E + 15) / 16;
    int blocks_e = (tiles_e + 7) / 8;
    int sums_total = B * 32 * N;

    // pack weights into WMMA A-operand images (L2-resident, 24KB)
    k_pack<<<1, 32, 0, stream>>>(cW1, cW2, vW1, vW2, eW1, eW2, aW1, aW2, oW1, wpk);

    // degree counts -> reciprocal counts
    k_zero<<<(N + T - 1) / T, T, 0, stream>>>(cnt, N);
    k_count<<<(E + T - 1) / T, T, 0, stream>>>(edges, cnt, E);
    k_invcnt<<<(N + T - 1) / T, T, 0, stream>>>(cnt, icnt, N);

    // per-node front: config -> vtx MLP -> v (34 ch, node-major f16)
    k_front<<<blocks_v, T, 0, stream>>>(verts, xW, xb, yW, yb, tW, tb,
                                        cb1, cb2, vb1, vb2, wpk, v_h, B, N);

    // 6 message-passing rounds
    for (int pass = 0; pass < 6; ++pass) {
        k_zero<<<(sums_total + T - 1) / T, T, 0, stream>>>(sums, sums_total);
        const _Float16* inbuf = (pass == 0) ? v_h : h_h;
        k_edge<<<blocks_e, T, 0, stream>>>(inbuf, edges, eb1, eb2, wpk,
                                           sums, B, N, E);
        k_finalize<<<(sums_total + T - 1) / T, T, 0, stream>>>(
            sums, icnt, nv, nvh, N, sums_total, pass == 0 ? 1 : 0);
        if (pass < 5)
            k_aft<<<blocks_v, T, 0, stream>>>(nvh, verts, ab1, ab2, wpk,
                                              h_h, B, N);
    }

    // output head
    k_vo<<<blocks_v, T, 0, stream>>>(nvh, ob1, oW2, ob2, wpk, fv, B, N);
    k_out<<<B, T, 0, stream>>>(fv, goW, gob, (float*)d_out, N);
}
